// BSDESolver_54348516163988
// MI455X (gfx1250) — matrix operand order repaired and checked by reading, not compiled, not run on hardware
//
#include <hip/hip_runtime.h>

#define BLOCK 64
#define NW    2      // waves per block
#define AROW  72     // padded A-row stride in halfs (144 B, 16B-aligned, bank-spread)
#define H2ROW 68     // padded h2-row stride in floats (272 B)

typedef __attribute__((ext_vector_type(16))) _Float16     v16h;
typedef __attribute__((ext_vector_type(8)))  float        v8f;
typedef __attribute__((ext_vector_type(4)))  unsigned int v4u;

union V16H { v16h h; v4u u[2]; };

__device__ __forceinline__ float gelu_exact(float x) {
    // jax.nn.gelu(approximate=False): 0.5*x*(1+erf(x/sqrt(2)))
    return 0.5f * x * (1.0f + erff(x * 0.70710678118654752f));
}

// Wave-local LDS ordering: LDS ops are in-order within a wave on CDNA5;
// the asm stops the compiler from hoisting loads above the stores.
__device__ __forceinline__ void lds_fence() {
    asm volatile("s_wait_dscnt 0" ::: "memory");
}

__global__ __launch_bounds__(BLOCK) void bsde_persistent_kernel(
    const float* __restrict__ dw,  const float* __restrict__ tg,
    const float* __restrict__ W1,  const float* __restrict__ b1,
    const float* __restrict__ g1,  const float* __restrict__ be1,
    const float* __restrict__ W2,  const float* __restrict__ b2,
    const float* __restrict__ g2,  const float* __restrict__ be2,
    const float* __restrict__ W3,  const float* __restrict__ b3,
    const float* __restrict__ Y0,  float* __restrict__ out, int Btot)
{
    __shared__ __align__(16) _Float16 aHi[NW * 32 * AROW];
    __shared__ __align__(16) _Float16 aLo[NW * 32 * AROW];
    __shared__ __align__(16) float    h2B[NW * 32 * H2ROW];
    __shared__ float sW1[128], sB1[64], sG1[64], sBe1[64],
                     sB2[64],  sG2[64], sBe2[64], sW3[64];

    const int tid  = threadIdx.x;
    const int wv   = tid >> 5;
    const int lane = tid & 31;
    const int gRow = blockIdx.x * BLOCK + tid;   // one BSDE path per lane

    for (int i = tid; i < 128; i += BLOCK) sW1[i] = W1[i];
    for (int i = tid; i < 64;  i += BLOCK) {
        sB1[i] = b1[i]; sG1[i] = g1[i]; sBe1[i] = be1[i];
        sB2[i] = b2[i]; sG2[i] = g2[i]; sBe2[i] = be2[i]; sW3[i] = W3[i];
    }
    __syncthreads();

    // ---- W2 B-fragments (32x16 K-major tiles), hi/lo f16 split, resident in VGPRs ----
    // B order used here (16-bit, 32x16): col N = lane%16; lanes 0-15 hold K=0..15, lanes 16-31 K=16..31 (contiguous, same as A).
    const int colN  = lane & 15;
    const int kHalf = lane >> 4;     // 0 or 1
    V16H bH[4][2], bL[4][2];
    #pragma unroll
    for (int nt = 0; nt < 4; ++nt)
        #pragma unroll
        for (int kc = 0; kc < 2; ++kc)
            #pragma unroll
            for (int e = 0; e < 16; ++e) {
                int k   = kc * 32 + kHalf * 16 + e;
                float w = W2[k * 64 + nt * 16 + colN];      // W2 is [K=64][N=64] row-major
                _Float16 hi = (_Float16)w;
                bH[nt][kc].h[e] = hi;
                bL[nt][kc].h[e] = (_Float16)(w - (float)hi);
            }

    float S = 100.0f;           // S0
    float Y = Y0[0];
    const float b3v = b3[0];
    const float RDT = 0.05f * 0.02f;   // R*DT
    const float SIG = 0.2f;            // SIGMA

    _Float16* myAH = &aHi[(wv * 32 + lane) * AROW];
    _Float16* myAL = &aLo[(wv * 32 + lane) * AROW];
    float*    myH2 = &h2B[(wv * 32 + lane) * H2ROW];

    #pragma unroll 1
    for (int step = 0; step < 50; ++step) {
        const float dwi = dw[gRow * 50 + step];
        const float ti  = tg[gRow * 50 + step];
        const float x0  = S * 0.01f;   // S / S0

        // ---------- layer 1: [x0,ti] @ W1 + b1 -> LN -> GELU (lane-local) ----------
        float h[64];
        float sum = 0.f;
        #pragma unroll
        for (int j = 0; j < 64; ++j) {
            h[j] = fmaf(x0, sW1[j], fmaf(ti, sW1[64 + j], sB1[j]));
            sum += h[j];
        }
        float mean = sum * (1.f / 64.f);
        float var = 0.f;
        #pragma unroll
        for (int j = 0; j < 64; ++j) { float d = h[j] - mean; var = fmaf(d, d, var); }
        float inv = rsqrtf(var * (1.f / 64.f) + 1e-5f);

        alignas(16) _Float16 rh[64];
        alignas(16) _Float16 rl[64];
        #pragma unroll
        for (int j = 0; j < 64; ++j) {
            float v = gelu_exact(fmaf((h[j] - mean) * inv, sG1[j], sBe1[j]));
            _Float16 hi = (_Float16)v;
            rh[j] = hi;
            rl[j] = (_Float16)(v - (float)hi);
        }
        #pragma unroll
        for (int c = 0; c < 8; ++c) {
            *(v4u*)&myAH[c * 8] = ((v4u*)rh)[c];
            *(v4u*)&myAL[c * 8] = ((v4u*)rl)[c];
        }
        lds_fence();

        // ---------- layer 2: h1 @ W2 via WMMA, hi/lo split (AhBh + AhBl + AlBh) ----------
        // A order used here (16-bit, 16x32): lane%16 = row M; lanes 0-15 K=0..15, lanes 16-31 K=16..31 (contiguous, same as B).
        #pragma unroll
        for (int rt = 0; rt < 2; ++rt) {
            V16H aH[2], aL[2];
            #pragma unroll
            for (int kc = 0; kc < 2; ++kc) {
                const _Float16* pH = &aHi[(wv * 32 + rt * 16 + colN) * AROW + kc * 32 + kHalf * 16];
                const _Float16* pL = &aLo[(wv * 32 + rt * 16 + colN) * AROW + kc * 32 + kHalf * 16];
                aH[kc].u[0] = *(const v4u*)pH;
                aH[kc].u[1] = *(const v4u*)(pH + 8);
                aL[kc].u[0] = *(const v4u*)pL;
                aL[kc].u[1] = *(const v4u*)(pL + 8);
            }
            #pragma unroll
            for (int nt = 0; nt < 4; ++nt) {
                v8f acc = {};
                #pragma unroll
                for (int kc = 0; kc < 2; ++kc) {
                    acc = __builtin_amdgcn_wmma_f32_16x16x32_f16(false, aH[kc].h, false, bH[nt][kc].h,
                                                                 (short)0, acc, false, false);
                    acc = __builtin_amdgcn_wmma_f32_16x16x32_f16(false, aH[kc].h, false, bL[nt][kc].h,
                                                                 (short)0, acc, false, false);
                    acc = __builtin_amdgcn_wmma_f32_16x16x32_f16(false, aL[kc].h, false, bH[nt][kc].h,
                                                                 (short)0, acc, false, false);
                }
                // D layout: VGPR e -> row (e or e+8 by lane half), col = lane%16
                const float bias = sB2[nt * 16 + colN];
                #pragma unroll
                for (int e = 0; e < 8; ++e) {
                    int row = rt * 16 + ((lane < 16) ? e : e + 8);
                    h2B[(wv * 32 + row) * H2ROW + nt * 16 + colN] = acc[e] + bias;
                }
            }
        }
        lds_fence();

        // ---------- layer 2 LN -> GELU -> head dot(W3) -> sigmoid -> SDE update ----------
        float x2[64];
        #pragma unroll
        for (int c = 0; c < 16; ++c) {
            float4 t = *(const float4*)&myH2[c * 4];
            x2[c * 4 + 0] = t.x; x2[c * 4 + 1] = t.y;
            x2[c * 4 + 2] = t.z; x2[c * 4 + 3] = t.w;
        }
        sum = 0.f;
        #pragma unroll
        for (int j = 0; j < 64; ++j) sum += x2[j];
        mean = sum * (1.f / 64.f);
        var = 0.f;
        #pragma unroll
        for (int j = 0; j < 64; ++j) { float d = x2[j] - mean; var = fmaf(d, d, var); }
        inv = rsqrtf(var * (1.f / 64.f) + 1e-5f);

        float dot = 0.f;
        #pragma unroll
        for (int j = 0; j < 64; ++j) {
            float v = gelu_exact(fmaf((x2[j] - mean) * inv, sG2[j], sBe2[j]));
            dot = fmaf(v, sW3[j], dot);
        }
        const float zeta = 1.f / (1.f + expf(-(dot + b3v)));

        const float dS = S * fmaf(SIG, dwi, RDT);
        Y = Y + RDT * (Y - zeta * S) + zeta * dS;
        S = S + dS;
        lds_fence();   // keep next iter's aHi/aLo stores behind this iter's consumers
    }

    out[gRow]        = Y;   // Y output, [B,1] flattened first
    out[Btot + gRow] = S;   // S output second
}

extern "C" void kernel_launch(void* const* d_in, const int* in_sizes, int n_in,
                              void* d_out, int out_size, void* d_ws, size_t ws_size,
                              hipStream_t stream) {
    (void)n_in; (void)out_size; (void)d_ws; (void)ws_size;
    const float* dw  = (const float*)d_in[0];
    const float* tg  = (const float*)d_in[1];
    const float* W1  = (const float*)d_in[2];
    const float* b1  = (const float*)d_in[3];
    const float* g1  = (const float*)d_in[4];
    const float* be1 = (const float*)d_in[5];
    const float* W2  = (const float*)d_in[6];
    const float* b2  = (const float*)d_in[7];
    const float* g2  = (const float*)d_in[8];
    const float* be2 = (const float*)d_in[9];
    const float* W3  = (const float*)d_in[10];
    const float* b3  = (const float*)d_in[11];
    const float* Y0  = (const float*)d_in[12];
    float* out = (float*)d_out;

    const int B = in_sizes[0] / 50;      // 32768
    const int grid = B / BLOCK;          // 512 blocks of 64 threads (2 waves)
    bsde_persistent_kernel<<<grid, BLOCK, 0, stream>>>(
        dw, tg, W1, b1, g1, be1, W2, b2, g2, be2, W3, b3, Y0, out, B);
}
